// MultiHeadSelfAttention_54271206752893
// MI455X (gfx1250) — matrix operand order repaired and checked by reading, not compiled, not run on hardware
//
#include <hip/hip_runtime.h>
#include <hip/hip_bf16.h>

typedef __bf16 bfx4  __attribute__((ext_vector_type(4)));
typedef __bf16 bfx8  __attribute__((ext_vector_type(8)));
typedef __bf16 bfx16 __attribute__((ext_vector_type(16)));
typedef float  v8f   __attribute__((ext_vector_type(8)));
typedef float  fx4   __attribute__((ext_vector_type(4)));

#define DMODEL 1024
#define NHEAD  16
#define HDIM   64
#define SEQ    2048
#define NROWS  8192   // B*S = 4*2048

#define USE_ASYNC_LDS 1

// ---- helpers -------------------------------------------------------------

__device__ __forceinline__ __bf16 f2bf(float f) {
  union { float f; unsigned u; } x; x.f = f;
  unsigned r = (x.u + 0x7FFFu + ((x.u >> 16) & 1u)) >> 16;  // round-nearest-even
  unsigned short hs = (unsigned short)r;
  __bf16 o; __builtin_memcpy(&o, &hs, sizeof(o));
  return o;
}

// A-fragment (16x32 bf16, wave32): per-lane row = lane%16, 16 contiguous
// elems at k + half*16 (matches contiguous B fragments).
__device__ __forceinline__ bfx16 load_afrag(const __bf16* row, int k, int hl) {
  const __bf16* p0 = row + k + hl * 16;
  bfx8 lo = *(const bfx8*)p0;
  bfx8 hi = *(const bfx8*)(p0 + 8);
  return __builtin_shufflevector(lo, hi, 0,1,2,3,4,5,6,7,8,9,10,11,12,13,14,15);
}

__device__ __forceinline__ v8f wmma_bf16(bfx16 a, bfx16 b, v8f c) {
  return __builtin_amdgcn_wmma_f32_16x16x32_bf16(false, a, false, b,
                                                 (short)0, c, false, false);
}

// 16B global -> LDS async copy (per-lane addresses), tracked by ASYNCcnt.
// LDS byte offset = low 32 bits of the generic pointer (flat LDS aperture).
__device__ __forceinline__ void async_g2l_b128(void* lds_dst, const void* gsrc) {
#if USE_ASYNC_LDS
  unsigned loff = (unsigned)(unsigned long long)(uintptr_t)lds_dst;
  unsigned long long ga = (unsigned long long)(uintptr_t)gsrc;
  asm volatile("global_load_async_to_lds_b128 %0, %1, off"
               :: "v"(loff), "v"(ga) : "memory");
#else
  *(bfx8*)lds_dst = *(const bfx8*)gsrc;
#endif
}

__device__ __forceinline__ void wait_async0() {
#if USE_ASYNC_LDS
  asm volatile("s_wait_asynccnt 0x0" ::: "memory");
#endif
}

// ---- fp32 -> bf16 convert ------------------------------------------------

__global__ __launch_bounds__(256) void cvt_f32_bf16(const float* __restrict__ src,
                                                    __bf16* __restrict__ dst, int n4) {
  int i = blockIdx.x * blockDim.x + threadIdx.x;
  if (i >= n4) return;
  fx4 v = *(const fx4*)(src + i * 4);
  bfx4 o;
  o[0] = f2bf(v[0]); o[1] = f2bf(v[1]); o[2] = f2bf(v[2]); o[3] = f2bf(v[3]);
  *(bfx4*)(dst + i * 4) = o;
}

// ---- GEMM: Y = A[N,K] @ W[D,K]^T + bias ---------------------------------
// Block: 8 waves, tile M=256 (32 rows/wave) x N=64.
// W tile (64 cols x 32 k) is async-staged to LDS once per block, double
// buffered; all 8 waves consume it (8x less global B traffic).
// mode 0: bf16 head-split [B,H,S,hd] (Q,K)  mode 1: bf16 [B,H,hd,S] (V)
// mode 2: fp32 row-major [N,D] (output projection)

__global__ __launch_bounds__(256) void gemm_qkvo(const __bf16* __restrict__ A,
                                                 const __bf16* __restrict__ W,
                                                 const float*  __restrict__ bias,
                                                 void* __restrict__ out, int mode) {
  __shared__ __bf16 sB[2][64 * 32];       // [n][k], row stride 32 elems

  const int tid  = threadIdx.x;
  const int lane = tid & 31;
  const int wave = tid >> 5;
  const int hl   = lane >> 4;
  const int l16  = lane & 15;
  const int n0   = blockIdx.x * 64;
  const int m0   = blockIdx.y * 256 + wave * 32;

  // staging: 64 rows x 4 chunks of 16B -> one b128 per thread per k-step
  const int brow = tid >> 2;
  const int bch  = (tid & 3) * 8;

  v8f acc[2][4] = {};

  const __bf16* arow0 = A + (size_t)(m0 + l16) * DMODEL;
  const __bf16* arow1 = arow0 + (size_t)16 * DMODEL;
  const __bf16* wrow  = W + (size_t)(n0 + brow) * DMODEL + bch;

  // prologue: stage k=0 into buffer 0
  async_g2l_b128(&sB[0][brow * 32 + bch], wrow);
  wait_async0();
  __syncthreads();

  int cur = 0;
  for (int k = 0; k < DMODEL; k += 32) {
    if (k + 32 < DMODEL)   // stage next k-block behind this block's WMMAs
      async_g2l_b128(&sB[cur ^ 1][brow * 32 + bch], wrow + k + 32);

    bfx16 a0 = load_afrag(arow0, k, hl);
    bfx16 a1 = load_afrag(arow1, k, hl);
#pragma unroll
    for (int t = 0; t < 4; ++t) {
      bfx16 bfr = *(const bfx16*)&sB[cur][(t * 16 + l16) * 32 + hl * 16];
      acc[0][t] = wmma_bf16(a0, bfr, acc[0][t]);
      acc[1][t] = wmma_bf16(a1, bfr, acc[1][t]);
    }

    if (k + 32 < DMODEL) wait_async0();
    __syncthreads();
    cur ^= 1;
  }

#pragma unroll
  for (int u = 0; u < 2; ++u) {
#pragma unroll
    for (int t = 0; t < 4; ++t) {
      int j = n0 + t * 16 + l16;
      float bj = bias[j];
#pragma unroll
      for (int r = 0; r < 8; ++r) {
        int m = m0 + u * 16 + r + hl * 8;   // C-layout: M = r + half*8
        float v = acc[u][t][r] + bj;
        if (mode == 2) {
          ((float*)out)[(size_t)m * DMODEL + j] = v;
        } else {
          int bb = m >> 11;                 // / SEQ
          int ss = m & (SEQ - 1);
          int hh = j >> 6;                  // / HDIM
          int dd = j & (HDIM - 1);
          __bf16* o = (__bf16*)out;
          if (mode == 0)
            o[((size_t)(bb * NHEAD + hh) * SEQ + ss) * HDIM + dd] = f2bf(v);
          else
            o[((size_t)(bb * NHEAD + hh) * HDIM + dd) * SEQ + ss] = f2bf(v);
        }
      }
    }
  }
}

// ---- fused flash attention ----------------------------------------------
// Block: 8 waves, all on the same (b,h); wave owns a 16-query tile.
// Per 32-key block, K tile (32x64) and V^T tile (64x32) are async-staged
// to LDS once per block and shared by all 8 waves (8x less K/V traffic),
// double buffered behind the WMMAs.

__global__ __launch_bounds__(256) void attn_fused(const __bf16* __restrict__ Qb,
                                                  const __bf16* __restrict__ Kb,
                                                  const __bf16* __restrict__ Vt,
                                                  __bf16* __restrict__ Ctx) {
  __shared__ __bf16 sK[2][32 * 64];       // [key][hd],  row stride 64 elems
  __shared__ __bf16 sV[2][64 * 32];       // [hd][key],  row stride 32 elems
  __shared__ __bf16 sP[8][16 * 32];       // per-wave P staging

  const int tid  = threadIdx.x;
  const int lane = tid & 31;
  const int wave = tid >> 5;
  const int hl   = lane >> 4;
  const int l16  = lane & 15;

  const int bh = blockIdx.x;              // 0..63
  const int q0 = (blockIdx.y * 8 + wave) * 16;
  const int b  = bh >> 4;
  const int h  = bh & 15;

  const __bf16* Q = Qb + (size_t)bh * SEQ * HDIM;
  const __bf16* K = Kb + (size_t)bh * SEQ * HDIM;
  const __bf16* V = Vt + (size_t)bh * HDIM * SEQ;   // [hd, S]

  // Q A-fragments for this tile (hd split into two K=32 chunks)
  const __bf16* qrow = Q + (size_t)(q0 + l16) * HDIM;
  bfx16 aq0 = load_afrag(qrow, 0, hl);
  bfx16 aq1 = load_afrag(qrow, 32, hl);

  // staging: K = 32 rows x 8 chunks, V = 64 rows x 4 chunks (16B each)
  const int krow = tid >> 3, kch = (tid & 7) * 8;
  const int vrow = tid >> 2, vch = (tid & 3) * 8;
  const __bf16* kg = K + (size_t)krow * HDIM + kch;
  const __bf16* vg = V + (size_t)vrow * SEQ + vch;

  v8f acc[4] = {};
  float mrow[8], lrow[8];
#pragma unroll
  for (int r = 0; r < 8; ++r) { mrow[r] = -__builtin_inff(); lrow[r] = 0.f; }

  __bf16* pbuf = &sP[wave][0];

  // prologue: stage key block 0 into buffer 0
  async_g2l_b128(&sK[0][krow * 64 + kch], kg);
  async_g2l_b128(&sV[0][vrow * 32 + vch], vg);
  wait_async0();
  __syncthreads();

  int cur = 0;
  for (int kb = 0; kb < SEQ; kb += 32) {
    if (kb + 32 < SEQ) {
      async_g2l_b128(&sK[cur ^ 1][krow * 64 + kch], kg + (size_t)(kb + 32) * HDIM);
      async_g2l_b128(&sV[cur ^ 1][vrow * 32 + vch], vg + kb + 32);
    }

    // ---- scores: two 16x16 tiles over 32 keys (K from LDS)
    bfx16 b00 = *(const bfx16*)&sK[cur][(l16) * 64      + hl * 16];
    bfx16 b01 = *(const bfx16*)&sK[cur][(l16) * 64 + 32 + hl * 16];
    bfx16 b10 = *(const bfx16*)&sK[cur][(16 + l16) * 64      + hl * 16];
    bfx16 b11 = *(const bfx16*)&sK[cur][(16 + l16) * 64 + 32 + hl * 16];

    v8f s0 = {}, s1 = {};
    s0 = wmma_bf16(aq0, b00, s0);
    s0 = wmma_bf16(aq1, b01, s0);
    s1 = wmma_bf16(aq0, b10, s1);
    s1 = wmma_bf16(aq1, b11, s1);

    // ---- online softmax over this key block
#pragma unroll
    for (int r = 0; r < 8; ++r) {
      float a0 = s0[r] * 0.125f;          // 1/sqrt(64)
      float a1 = s1[r] * 0.125f;
      float x = fmaxf(a0, a1);
      x = fmaxf(x, __shfl_xor(x, 1, 16));
      x = fmaxf(x, __shfl_xor(x, 2, 16));
      x = fmaxf(x, __shfl_xor(x, 4, 16));
      x = fmaxf(x, __shfl_xor(x, 8, 16));
      float mnew = fmaxf(mrow[r], x);
      float p0 = __expf(a0 - mnew);
      float p1 = __expf(a1 - mnew);
      float rs = p0 + p1;
      rs += __shfl_xor(rs, 1, 16);
      rs += __shfl_xor(rs, 2, 16);
      rs += __shfl_xor(rs, 4, 16);
      rs += __shfl_xor(rs, 8, 16);
      float sc = __expf(mrow[r] - mnew);
      lrow[r] = lrow[r] * sc + rs;
      mrow[r] = mnew;
      acc[0][r] *= sc; acc[1][r] *= sc; acc[2][r] *= sc; acc[3][r] *= sc;
      // stage P (bf16) row-major [16 x 32]; C-layout: row = r + hl*8, col = n
      pbuf[(r + hl * 8) * 32 + l16]      = f2bf(p0);
      pbuf[(r + hl * 8) * 32 + 16 + l16] = f2bf(p1);
    }

    // ---- rebuild P as an A-fragment (same-wave DS ops are in order)
    bfx8 plo = *(const bfx8*)(pbuf + l16 * 32 + hl * 16);
    bfx8 phi = *(const bfx8*)(pbuf + l16 * 32 + hl * 16 + 8);
    bfx16 ap = __builtin_shufflevector(plo, phi,
                 0,1,2,3,4,5,6,7,8,9,10,11,12,13,14,15);

    // ---- O += P @ V  (V^T tile from LDS: contiguous key runs per hd row)
#pragma unroll
    for (int t = 0; t < 4; ++t) {
      bfx16 bv = *(const bfx16*)&sV[cur][(t * 16 + l16) * 32 + hl * 16];
      acc[t] = wmma_bf16(ap, bv, acc[t]);
    }

    if (kb + 32 < SEQ) wait_async0();
    __syncthreads();
    cur ^= 1;
  }

  // ---- epilogue: O /= l, store context bf16 row-major [B*S, D]
#pragma unroll
  for (int r = 0; r < 8; ++r) {
    float inv = 1.0f / lrow[r];
    int m = q0 + r + hl * 8;
    size_t rowo = ((size_t)b * SEQ + m) * DMODEL + (size_t)h * HDIM;
#pragma unroll
    for (int t = 0; t < 4; ++t)
      Ctx[rowo + t * 16 + l16] = f2bf(acc[t][r] * inv);
  }
}

// ---- launcher ------------------------------------------------------------

extern "C" void kernel_launch(void* const* d_in, const int* in_sizes, int n_in,
                              void* d_out, int out_size, void* d_ws, size_t ws_size,
                              hipStream_t stream) {
  const float* x  = (const float*)d_in[0];
  const float* wq = (const float*)d_in[1];
  const float* bq = (const float*)d_in[2];
  const float* wk = (const float*)d_in[3];
  const float* bk = (const float*)d_in[4];
  const float* wv = (const float*)d_in[5];
  const float* bv = (const float*)d_in[6];
  const float* wo = (const float*)d_in[7];
  const float* bo = (const float*)d_in[8];

  char* ws = (char*)d_ws;
  const size_t XN = (size_t)NROWS * DMODEL;     // 8388608 elems
  const size_t WN = (size_t)DMODEL * DMODEL;    // 1048576 elems
  __bf16* Xb = (__bf16*)(ws);                          // 16 MiB
  __bf16* Wq = (__bf16*)(ws + XN * 2);                 // 2 MiB each
  __bf16* Wk = Wq + WN;
  __bf16* Wv = Wk + WN;
  __bf16* Wo = Wv + WN;
  __bf16* Qb = Wo + WN;                                // 16 MiB each
  __bf16* Kb = Qb + XN;
  __bf16* Vt = Kb + XN;
  __bf16* Cx = Vt + XN;

  dim3 blk(256);

  // fp32 -> bf16 conversions
  cvt_f32_bf16<<<(int)(XN / 4 / 256), blk, 0, stream>>>(x, Xb, (int)(XN / 4));
  cvt_f32_bf16<<<(int)(WN / 4 / 256), blk, 0, stream>>>(wq, Wq, (int)(WN / 4));
  cvt_f32_bf16<<<(int)(WN / 4 / 256), blk, 0, stream>>>(wk, Wk, (int)(WN / 4));
  cvt_f32_bf16<<<(int)(WN / 4 / 256), blk, 0, stream>>>(wv, Wv, (int)(WN / 4));
  cvt_f32_bf16<<<(int)(WN / 4 / 256), blk, 0, stream>>>(wo, Wo, (int)(WN / 4));

  // QKV projections: block tile 256x64
  dim3 gg(DMODEL / 64, NROWS / 256);   // (16, 32)
  gemm_qkvo<<<gg, blk, 0, stream>>>(Xb, Wq, bq, Qb, 0);
  gemm_qkvo<<<gg, blk, 0, stream>>>(Xb, Wk, bk, Kb, 0);
  gemm_qkvo<<<gg, blk, 0, stream>>>(Xb, Wv, bv, Vt, 1);

  // fused flash attention: grid = (B*H, S/128), 8 q-tiles per block
  attn_fused<<<dim3(64, 16), blk, 0, stream>>>(Qb, Kb, Vt, Cx);

  // output projection -> fp32 d_out
  gemm_qkvo<<<gg, blk, 0, stream>>>(Cx, Wo, bo, d_out, 2);
}
